// EncoderBase_87986700026432
// MI455X (gfx1250) — matrix operand order repaired and checked by reading, not compiled, not run on hardware
//
#include <hip/hip_runtime.h>

// ---------------------------------------------------------------------------
// Multi-hot embedding bag as f16 WMMA GEMM:
//   C[8192, 256] = X[8192, 8192](0/1 int32) * E[8192, 256](f32)
//   out = C * sqrt(256) + pos_enc[s,:] * (row has any token)
//
// MI455X roofline: x stream = 256 MB read-once -> ~11us HBM floor at 23.3TB/s;
// f16 WMMA compute ~7us (f32 WMMA would be ~20x compute-bound; fp8 would
// quantize the table for no gain since we're HBM bound on x). So: exact
// 0/1 -> f16 A via bit trick, f32 -> f16 B (one rounding), f32 accumulation.
//
// Cache policy: x / partials / out are streaming (read-or-write-once) -> NT
// hints so the 192MB L2 stays owned by the 4MB f16 table (re-read 16x per
// K-step by every wave) instead of being churned by the 256MB x stream.
// ---------------------------------------------------------------------------

typedef __attribute__((ext_vector_type(16))) _Float16 v16h;
typedef __attribute__((ext_vector_type(8)))  float    v8f;
typedef __attribute__((ext_vector_type(4)))  int      v4i;

#define B_      8
#define S_      1024
#define V_      8192
#define D_      256
#define M_      (B_ * S_)        // 8192 rows
#define KSPLIT  4
#define KCHUNK  (V_ / KSPLIT)    // 2048
#define NT_     (D_ / 16)        // 16 N tiles per wave

// ---------------------------------------------------------------------------
// Kernel 1: pack E f32 [V][D] -> f16 transposed Et [D][V].
// Writes coalesced over v; reads hit L2 after first touch (table is 8 MB,
// L2 is 192 MB). Et is written RT so it stays L2-resident for kernel 2.
// ---------------------------------------------------------------------------
__global__ void pack_emb_f16t(const float* __restrict__ emb,
                              _Float16* __restrict__ et) {
  int tid = blockIdx.x * blockDim.x + threadIdx.x;   // 0 .. V*D-1
  int d = tid >> 13;            // / 8192
  int v = tid & (V_ - 1);
  et[(size_t)d * V_ + v] = (_Float16)emb[(size_t)v * D_ + d];
}

__device__ __forceinline__ unsigned pack2h01(int a, int b) {
  // a,b in {0,1}: f16(1.0) = 0x3C00, f16(0.0) = 0x0000. Exact, no v_cvt chain.
  return (unsigned)(a * 0x3C00) | ((unsigned)(b * 0x3C00) << 16);
}

// ---------------------------------------------------------------------------
// Kernel 2: WMMA GEMM. One wave = 16-row M tile x full 256 cols, one K chunk.
// 2048 waves = 256 blocks * 8 waves. x read exactly once from HBM (NT).
// A layout (16-bit A 16x32): lane l (half=l>>4, r=l&15) holds row r,
//   elems 0..7  -> K = kb + half*16 + {0..7}
//   elems 8..15 -> K = kb + half*16 + 8 + {0..7}
// B layout (16-bit B 32x16): lane l holds col r, elem e -> K = kb + half*16 + e
// C layout: elem v of v8f, lane l -> row v + 8*half, col r.
// ---------------------------------------------------------------------------
__global__ void __launch_bounds__(256)
gemm_wmma(const int* __restrict__ x,
          const _Float16* __restrict__ et,
          float* __restrict__ part,        // [KSPLIT][M_][D_]
          int* __restrict__ anypart) {     // [KSPLIT][M_]
  const int lane   = threadIdx.x & 31;
  const int wave   = blockIdx.x * (blockDim.x >> 5) + (threadIdx.x >> 5);
  const int mtile  = wave >> 2;            // 0..511
  const int kchunk = wave & (KSPLIT - 1);  // 0..3
  const int half   = lane >> 4;
  const int r      = lane & 15;
  const long grow  = (long)mtile * 16 + r;

  const int* __restrict__ xrow = x + grow * (long)V_ + (long)kchunk * KCHUNK;
  const _Float16* __restrict__ etk = et + (long)kchunk * KCHUNK;

  v8f acc[NT_];
#pragma unroll
  for (int i = 0; i < NT_; ++i) acc[i] = (v8f){0, 0, 0, 0, 0, 0, 0, 0};

  unsigned anyacc = 0;

  for (int kb = 0; kb < KCHUNK; kb += 32) {
    // ---- A fragment: two contiguous 32B int32 chunks per lane (NT: x is
    // a 256MB read-once stream; keep it out of L2) ----
    const int* pa0 = xrow + kb + half * 16;
    const int* pa1 = pa0 + 8;
    v4i xa = __builtin_nontemporal_load((const v4i*)(pa0));
    v4i xb = __builtin_nontemporal_load((const v4i*)(pa0 + 4));
    v4i xc = __builtin_nontemporal_load((const v4i*)(pa1));
    v4i xd = __builtin_nontemporal_load((const v4i*)(pa1 + 4));
    anyacc |= (unsigned)(xa.x | xa.y | xa.z | xa.w | xb.x | xb.y | xb.z | xb.w |
                         xc.x | xc.y | xc.z | xc.w | xd.x | xd.y | xd.z | xd.w);

    union { v16h h; unsigned u[8]; } A;
    A.u[0] = pack2h01(xa.x, xa.y);
    A.u[1] = pack2h01(xa.z, xa.w);
    A.u[2] = pack2h01(xb.x, xb.y);
    A.u[3] = pack2h01(xb.z, xb.w);
    A.u[4] = pack2h01(xc.x, xc.y);
    A.u[5] = pack2h01(xc.z, xc.w);
    A.u[6] = pack2h01(xd.x, xd.y);
    A.u[7] = pack2h01(xd.z, xd.w);

    // ---- 16 N tiles: B fragment (32B contiguous per lane, L2-hot, RT) ----
#pragma unroll
    for (int nt = 0; nt < NT_; ++nt) {
      const _Float16* pb = etk + (long)(nt * 16 + r) * V_ + kb + half * 16;
      v16h b = *(const v16h*)pb;
      acc[nt] = __builtin_amdgcn_wmma_f32_16x16x32_f16(
          /*neg_a=*/false, A.h, /*neg_b=*/false, b,
          /*c_mod=*/(short)0, acc[nt], /*reuse_a=*/false, /*reuse_b=*/false);
    }
  }

  // ---- row-any flag: combine the two K-halves held by lanes l and l^16 ----
  unsigned rowany = anyacc | (unsigned)__shfl_xor((int)anyacc, 16, 32);
  if (half == 0) {
    anypart[(long)kchunk * M_ + grow] = (int)rowany;
  }

  // ---- store partials (write-once, consumed by kernel 3: NT) ----
  float* pbase = part + (long)kchunk * M_ * D_;
#pragma unroll
  for (int nt = 0; nt < NT_; ++nt) {
#pragma unroll
    for (int vv = 0; vv < 8; ++vv) {
      long row = (long)mtile * 16 + vv + 8 * half;
      __builtin_nontemporal_store(acc[nt][vv],
                                  &pbase[row * D_ + nt * 16 + r]);
    }
  }
}

// ---------------------------------------------------------------------------
// Kernel 3: reduce K-split partials, scale by sqrt(D)=16, gated pos_enc add.
// Partials are read-once (NT); out is write-once (NT); pos_enc is hot (RT).
// ---------------------------------------------------------------------------
__global__ void reduce_epilogue(const float* __restrict__ part,
                                const int* __restrict__ anypart,
                                const float* __restrict__ pos_enc,
                                float* __restrict__ out) {
  long idx = (long)blockIdx.x * blockDim.x + threadIdx.x;  // 0 .. M*D-1
  long row = idx >> 8;           // / 256
  int d = (int)(idx & (D_ - 1));
  int s = (int)(row & (S_ - 1));

  const long PSZ = (long)M_ * D_;
  float sum = __builtin_nontemporal_load(&part[idx]) +
              __builtin_nontemporal_load(&part[PSZ + idx]) +
              __builtin_nontemporal_load(&part[2 * PSZ + idx]) +
              __builtin_nontemporal_load(&part[3 * PSZ + idx]);
  int any = anypart[row] | anypart[M_ + row] |
            anypart[2 * M_ + row] | anypart[3 * M_ + row];

  float o = sum * 16.0f;                       // sqrt(256)
  if (any) o += pos_enc[(long)s * D_ + d];
  __builtin_nontemporal_store(o, &out[idx]);
}

// ---------------------------------------------------------------------------
extern "C" void kernel_launch(void* const* d_in, const int* in_sizes, int n_in,
                              void* d_out, int out_size, void* d_ws, size_t ws_size,
                              hipStream_t stream) {
  const int*   x   = (const int*)d_in[0];     // [B,S,V] int32 0/1
  const float* emb = (const float*)d_in[1];   // [V,D] f32
  const float* pos = (const float*)d_in[2];   // [1,MAX_POS,D] f32
  float*       out = (float*)d_out;           // [B,S,D] f32

  // workspace layout
  char* ws = (char*)d_ws;
  _Float16* et   = (_Float16*)ws;                                   // 4 MB
  float*    part = (float*)(ws + (size_t)V_ * D_ * sizeof(_Float16));        // 32 MB
  int*      anyp = (int*)(ws + (size_t)V_ * D_ * sizeof(_Float16)
                             + (size_t)KSPLIT * M_ * D_ * sizeof(float));    // 128 KB

  pack_emb_f16t<<<(V_ * D_) / 256, 256, 0, stream>>>(emb, et);
  gemm_wmma<<<(M_ / 16) * KSPLIT / 8, 256, 0, stream>>>(x, et, part, anyp);
  reduce_epilogue<<<(M_ * D_) / 256, 256, 0, stream>>>(part, anyp, pos, out);
}
